// GraphAttentionEmbedding_78176994721830
// MI455X (gfx1250) — compile-verified, matrix operand order checked by reading, not run on hardware
//
#include <hip/hip_runtime.h>
#include <cstdint>

#define N_NODES 100000
#define N_EDGES 640000
#define DIM     128
#define EDIM    256

typedef _Float16 half_t;
typedef __attribute__((ext_vector_type(16))) _Float16 v16h;
typedef __attribute__((ext_vector_type(8)))  float   v8f;

union FragH { uint4 u[2]; v16h v; };
union Pack8 { _Float16 h[8]; uint4 u; };
union Pack4 { _Float16 h[4]; uint2 u; };

__device__ __forceinline__ v8f splat8(float x) {
    v8f r = {x, x, x, x, x, x, x, x};
    return r;
}

// ---------------------------------------------------------------------------
// Vectorized f32 -> f16 convert (x), 4 elements per thread
// ---------------------------------------------------------------------------
__global__ void cvt_f16_kernel(const float4* __restrict__ in, uint2* __restrict__ out, int n4) {
    int i = blockIdx.x * blockDim.x + threadIdx.x;
    if (i >= n4) return;
    float4 f = in[i];
    Pack4 p;
    p.h[0] = (half_t)f.x; p.h[1] = (half_t)f.y;
    p.h[2] = (half_t)f.z; p.h[3] = (half_t)f.w;
    out[i] = p.u;
}

// ---------------------------------------------------------------------------
// Pack row-major W[K,128] (f32) into CDNA5 WMMA B-fragment layout (f16).
// idx = ((nt*KS + ks)*32 + lane)*16 + i
//   l = lane&15 (output col in tile), hi = lane>>4
//   i -> dword v=i/2, half b=i&1 ; k_local = (v<4 ? 2v+b : 16+2(v-4)+b) + 8*hi
// ---------------------------------------------------------------------------
__global__ void pack_weight_kernel(const float* __restrict__ W, half_t* __restrict__ Wp, int KS) {
    int idx = blockIdx.x * blockDim.x + threadIdx.x;
    int total = 8 * KS * 32 * 16;
    if (idx >= total) return;
    int i   = idx & 15;
    int L   = (idx >> 4) & 31;
    int r2  = idx >> 9;
    int ks  = r2 % KS;
    int nt  = r2 / KS;
    int l   = L & 15, hi = L >> 4;
    int v   = i >> 1, b = i & 1;
    int klo = (v < 4) ? (2 * v + b) : (16 + 2 * (v - 4) + b);
    int k   = ks * 32 + 8 * hi + klo;
    int n   = nt * 16 + l;
    Wp[idx] = (half_t)W[k * DIM + n];
}

__global__ void init_stats_kernel(unsigned int* __restrict__ nmax, float* __restrict__ nsum, int n) {
    int i = blockIdx.x * blockDim.x + threadIdx.x;
    if (i < n) { nmax[i] = 0u; nsum[i] = 0.0f; }
}

// ---------------------------------------------------------------------------
// Node projection GEMM: out[N,128] = xb @ W + bias. One wave per 16x128 stripe.
// A fragments preloaded for all 4 k-steps; B fragments batched per k-step.
// ---------------------------------------------------------------------------
template <typename OutT>
__global__ __launch_bounds__(256)
void node_gemm_kernel(const half_t* __restrict__ xb, const half_t* __restrict__ Wp,
                      const float* __restrict__ bias, OutT* __restrict__ out) {
    const int lane = threadIdx.x & 31;
    const int wid  = threadIdx.x >> 5;
    const int tile = blockIdx.x * 8 + wid;
    if (tile >= N_NODES / 16) return;          // wave-uniform guard
    const int l = lane & 15, hi = lane >> 4;
    const int row0 = tile * 16;

    const half_t* arow = xb + (size_t)(row0 + l) * DIM;

    FragH a[4];
#pragma unroll
    for (int ks = 0; ks < 4; ++ks) {
        a[ks].u[0] = *(const uint4*)(arow + ks * 32 + 8 * hi);
        a[ks].u[1] = *(const uint4*)(arow + ks * 32 + 8 * hi + 16);
    }

    v8f acc[8];
#pragma unroll
    for (int nt = 0; nt < 8; ++nt) acc[nt] = splat8(bias[nt * 16 + l]);

#pragma unroll
    for (int ks = 0; ks < 4; ++ks) {
        FragH b[8];
#pragma unroll
        for (int nt = 0; nt < 8; ++nt) {
            const half_t* bp = Wp + ((size_t)(nt * 4 + ks) * 32 + lane) * 16;
            b[nt].u[0] = *(const uint4*)(bp);
            b[nt].u[1] = *(const uint4*)(bp + 8);
        }
#pragma unroll
        for (int nt = 0; nt < 8; ++nt)
            acc[nt] = __builtin_amdgcn_wmma_f32_16x16x32_f16(
                false, a[ks].v, false, b[nt].v, (short)0, acc[nt], false, false);
    }
#pragma unroll
    for (int nt = 0; nt < 8; ++nt)
#pragma unroll
        for (int r = 0; r < 8; ++r)
            out[(size_t)(row0 + r + 8 * hi) * DIM + nt * 16 + l] = (OutT)acc[nt][r];
}

// ---------------------------------------------------------------------------
// Edge projection GEMM: eproj[E,128] (f16) = [cos(dt*w+b) || msg] @ We[256,128]
// Each wave builds its 16x256 f16 A-tile in LDS (b128 stores), 8 WMMA k-steps.
// ---------------------------------------------------------------------------
__global__ __launch_bounds__(128)
void edge_gemm_kernel(const half_t* __restrict__ Wep,
                      const float* __restrict__ t, const float* __restrict__ last_update,
                      const float* __restrict__ msg,
                      const float* __restrict__ tw, const float* __restrict__ tb,
                      const long long* __restrict__ ei, half_t* __restrict__ eproj) {
    __shared__ __align__(16) half_t tileA[4][16 * EDIM];
    const int lane = threadIdx.x & 31;
    const int wid  = threadIdx.x >> 5;
    const int tl   = blockIdx.x * 4 + wid;
    const bool active = (tl < N_EDGES / 16);   // wave-uniform
    half_t* my = tileA[wid];

    if (active) {
        const int e0 = tl * 16;
        if (lane < 16) {
            const int j0 = lane * 8;           // time-encoding cols [j0, j0+8)
            float tww[8], tbb[8];
#pragma unroll
            for (int jj = 0; jj < 8; ++jj) { tww[jj] = tw[j0 + jj]; tbb[jj] = tb[j0 + jj]; }
            for (int r = 0; r < 16; ++r) {
                int e = e0 + r;
                long long s = ei[e];
                float rel = t[e] - last_update[s];
                Pack8 p;
#pragma unroll
                for (int jj = 0; jj < 8; ++jj)
                    p.h[jj] = (half_t)__builtin_cosf(rel * tww[jj] + tbb[jj]);
                *(uint4*)(my + r * EDIM + j0) = p.u;
            }
        } else {
            const int c0 = (lane - 16) * 8;    // message cols [128+c0, 128+c0+8)
            for (int r = 0; r < 16; ++r) {
                const float* mrow = msg + (size_t)(e0 + r) * DIM + c0;
                float4 m0 = *(const float4*)(mrow);
                float4 m1 = *(const float4*)(mrow + 4);
                Pack8 p;
                p.h[0] = (half_t)m0.x; p.h[1] = (half_t)m0.y;
                p.h[2] = (half_t)m0.z; p.h[3] = (half_t)m0.w;
                p.h[4] = (half_t)m1.x; p.h[5] = (half_t)m1.y;
                p.h[6] = (half_t)m1.z; p.h[7] = (half_t)m1.w;
                *(uint4*)(my + r * EDIM + 128 + c0) = p.u;
            }
        }
    }
    __syncthreads();
    if (!active) return;

    const int l = lane & 15, hi = lane >> 4;
    const int e0 = tl * 16;
    const half_t* arow = my + l * EDIM;

    v8f acc[8];
#pragma unroll
    for (int nt = 0; nt < 8; ++nt) acc[nt] = splat8(0.0f);

#pragma unroll
    for (int ks = 0; ks < 8; ++ks) {
        FragH a;
        a.u[0] = *(const uint4*)(arow + ks * 32 + 8 * hi);
        a.u[1] = *(const uint4*)(arow + ks * 32 + 8 * hi + 16);
        FragH b[8];
#pragma unroll
        for (int nt = 0; nt < 8; ++nt) {
            const half_t* bp = Wep + ((size_t)(nt * 8 + ks) * 32 + lane) * 16;
            b[nt].u[0] = *(const uint4*)(bp);
            b[nt].u[1] = *(const uint4*)(bp + 8);
        }
#pragma unroll
        for (int nt = 0; nt < 8; ++nt)
            acc[nt] = __builtin_amdgcn_wmma_f32_16x16x32_f16(
                false, a.v, false, b[nt].v, (short)0, acc[nt], false, false);
    }
#pragma unroll
    for (int nt = 0; nt < 8; ++nt)
#pragma unroll
        for (int r = 0; r < 8; ++r)
            eproj[(size_t)(e0 + r + 8 * hi) * DIM + nt * 16 + l] = (half_t)acc[nt][r];
}

// ---------------------------------------------------------------------------
// Per-edge attention logits + scatter max. One wave per edge; lanes 0..15 are
// head 0 (c = lane*4..+3), lanes 16..31 head 1.
// ---------------------------------------------------------------------------
__global__ __launch_bounds__(256)
void alpha_kernel(const half_t* __restrict__ qn, const half_t* __restrict__ kn,
                  const half_t* __restrict__ eproj, const long long* __restrict__ ei,
                  float* __restrict__ alpha, unsigned int* __restrict__ nmax) {
    const int lane = threadIdx.x & 31;
    const int e = blockIdx.x * 8 + (threadIdx.x >> 5);
    if (e >= N_EDGES) return;
    const long long s = ei[e], d = ei[N_EDGES + e];
    Pack4 q, k, ep;
    q.u  = *(const uint2*)(qn    + (size_t)d * DIM + lane * 4);
    k.u  = *(const uint2*)(kn    + (size_t)s * DIM + lane * 4);
    ep.u = *(const uint2*)(eproj + (size_t)e * DIM + lane * 4);
    float p = 0.0f;
#pragma unroll
    for (int j = 0; j < 4; ++j)
        p += (float)q.h[j] * ((float)k.h[j] + (float)ep.h[j]);
    p += __shfl_xor(p, 1, 32);
    p += __shfl_xor(p, 2, 32);
    p += __shfl_xor(p, 4, 32);
    p += __shfl_xor(p, 8, 32);           // sum within each 16-lane head group
    float a = p * 0.125f;                // 1/sqrt(64)
    if ((lane & 15) == 0) {
        int h = lane >> 4;
        alpha[(size_t)e * 2 + h] = a;
        unsigned int u = __float_as_uint(a);
        unsigned int key = (u & 0x80000000u) ? ~u : (u | 0x80000000u);
        atomicMax(&nmax[(size_t)d * 2 + h], key);
    }
}

// ---------------------------------------------------------------------------
// exp(alpha - max[dst]) + scatter sum
// ---------------------------------------------------------------------------
__global__ void exp_kernel(const float* __restrict__ alpha, const unsigned int* __restrict__ nmax,
                           const long long* __restrict__ ei,
                           float* __restrict__ aexp, float* __restrict__ nsum) {
    int idx = blockIdx.x * blockDim.x + threadIdx.x;
    if (idx >= N_EDGES * 2) return;
    int e = idx >> 1, h = idx & 1;
    long long d = ei[N_EDGES + e];
    unsigned int key  = nmax[d * 2 + h];
    unsigned int bits = (key & 0x80000000u) ? (key & 0x7FFFFFFFu) : ~key;
    float m  = __uint_as_float(bits);
    float ex = __expf(alpha[idx] - m);
    aexp[idx] = ex;
    atomicAdd(&nsum[d * 2 + h], ex);
}

// ---------------------------------------------------------------------------
// out[dst] += attn * (v_n[src] + e_proj)   (d_out pre-filled with skip)
// ---------------------------------------------------------------------------
__global__ __launch_bounds__(256)
void scatter_kernel(const half_t* __restrict__ vn, const half_t* __restrict__ eproj,
                    const float* __restrict__ aexp, const float* __restrict__ nsum,
                    const long long* __restrict__ ei, float* __restrict__ out) {
    const int lane = threadIdx.x & 31;
    const int e = blockIdx.x * 8 + (threadIdx.x >> 5);
    if (e >= N_EDGES) return;
    const long long s = ei[e], d = ei[N_EDGES + e];
    const int h = lane >> 4;
    float attn = aexp[(size_t)e * 2 + h] / (nsum[d * 2 + h] + 1e-16f);
    Pack4 v, ep;
    v.u  = *(const uint2*)(vn    + (size_t)s * DIM + lane * 4);
    ep.u = *(const uint2*)(eproj + (size_t)e * DIM + lane * 4);
    float* o = out + (size_t)d * DIM + lane * 4;
#pragma unroll
    for (int j = 0; j < 4; ++j)
        atomicAdd(o + j, attn * ((float)v.h[j] + (float)ep.h[j]));
}

// ---------------------------------------------------------------------------
extern "C" void kernel_launch(void* const* d_in, const int* in_sizes, int n_in,
                              void* d_out, int out_size, void* d_ws, size_t ws_size,
                              hipStream_t stream) {
    (void)in_sizes; (void)n_in; (void)out_size; (void)ws_size;
    const float*     x           = (const float*)d_in[0];
    const float*     last_update = (const float*)d_in[1];
    const float*     t           = (const float*)d_in[2];
    const float*     msg         = (const float*)d_in[3];
    const float*     time_w      = (const float*)d_in[4];
    const float*     time_b      = (const float*)d_in[5];
    const float*     Wq          = (const float*)d_in[6];
    const float*     bq          = (const float*)d_in[7];
    const float*     Wk          = (const float*)d_in[8];
    const float*     bk          = (const float*)d_in[9];
    const float*     Wv          = (const float*)d_in[10];
    const float*     bv          = (const float*)d_in[11];
    const float*     We          = (const float*)d_in[12];
    const float*     Wskip       = (const float*)d_in[13];
    const float*     bskip       = (const float*)d_in[14];
    const long long* ei          = (const long long*)d_in[15];
    float* out = (float*)d_out;

    // -------- workspace carve-up --------
    char* ws = (char*)d_ws;
    size_t off = 0;
    auto carve = [&](size_t bytes) -> char* {
        off = (off + 255) & ~(size_t)255;
        char* p = ws + off;
        off += bytes;
        return p;
    };
    half_t*       xb    = (half_t*)carve((size_t)N_NODES * DIM * 2);
    half_t*       Wqp   = (half_t*)carve((size_t)DIM * DIM * 2);
    half_t*       Wkp   = (half_t*)carve((size_t)DIM * DIM * 2);
    half_t*       Wvp   = (half_t*)carve((size_t)DIM * DIM * 2);
    half_t*       Wsp   = (half_t*)carve((size_t)DIM * DIM * 2);
    half_t*       Wep   = (half_t*)carve((size_t)EDIM * DIM * 2);
    half_t*       qn    = (half_t*)carve((size_t)N_NODES * DIM * 2);
    half_t*       kn    = (half_t*)carve((size_t)N_NODES * DIM * 2);
    half_t*       vn    = (half_t*)carve((size_t)N_NODES * DIM * 2);
    half_t*       eproj = (half_t*)carve((size_t)N_EDGES * DIM * 2);
    float*        alpha = (float*)carve((size_t)N_EDGES * 2 * 4);
    float*        aexp  = (float*)carve((size_t)N_EDGES * 2 * 4);
    unsigned int* nmax  = (unsigned int*)carve((size_t)N_NODES * 2 * 4);
    float*        nsum  = (float*)carve((size_t)N_NODES * 2 * 4);

    // 1) x -> f16 (vectorized)
    {
        int n4 = N_NODES * DIM / 4;
        cvt_f16_kernel<<<(n4 + 255) / 256, 256, 0, stream>>>(
            (const float4*)x, (uint2*)xb, n4);
    }
    // 2) pack weights into WMMA B-fragment layout
    pack_weight_kernel<<<64,  256, 0, stream>>>(Wq,    Wqp, 4);
    pack_weight_kernel<<<64,  256, 0, stream>>>(Wk,    Wkp, 4);
    pack_weight_kernel<<<64,  256, 0, stream>>>(Wv,    Wvp, 4);
    pack_weight_kernel<<<64,  256, 0, stream>>>(Wskip, Wsp, 4);
    pack_weight_kernel<<<128, 256, 0, stream>>>(We,    Wep, 8);
    // 3) init scatter stats
    init_stats_kernel<<<(N_NODES * 2 + 255) / 256, 256, 0, stream>>>(nmax, nsum, N_NODES * 2);
    // 4) node projections (skip goes straight into d_out as f32)
    {
        dim3 grid((N_NODES / 16 + 7) / 8);
        node_gemm_kernel<half_t><<<grid, 256, 0, stream>>>(xb, Wqp, bq,    qn);
        node_gemm_kernel<half_t><<<grid, 256, 0, stream>>>(xb, Wkp, bk,    kn);
        node_gemm_kernel<half_t><<<grid, 256, 0, stream>>>(xb, Wvp, bv,    vn);
        node_gemm_kernel<float> <<<grid, 256, 0, stream>>>(xb, Wsp, bskip, out);
    }
    // 5) edge projection GEMM ([cos||msg] tile in LDS per wave)
    edge_gemm_kernel<<<N_EDGES / 16 / 4, 128, 0, stream>>>(
        Wep, t, last_update, msg, time_w, time_b, ei, eproj);
    // 6) attention logits + scatter max
    alpha_kernel<<<N_EDGES / 8, 256, 0, stream>>>(qn, kn, eproj, ei, alpha, nmax);
    // 7) exp + scatter sum
    exp_kernel<<<(N_EDGES * 2 + 255) / 256, 256, 0, stream>>>(alpha, nmax, ei, aexp, nsum);
    // 8) weighted scatter into d_out (on top of skip)
    scatter_kernel<<<N_EDGES / 8, 256, 0, stream>>>(vn, eproj, aexp, nsum, ei, out);
}
